// PointnetFPModule_57793079935589
// MI455X (gfx1250) — matrix operand order repaired and checked by reading, not compiled, not run on hardware
//
#include <hip/hip_runtime.h>
#include <stdint.h>

typedef __attribute__((ext_vector_type(16))) __bf16 v16bf;
typedef __attribute__((ext_vector_type(8)))  float  v8f;

// ---------------- problem constants ----------------
// B=4, N=16384, M=4096, C1=128, C2=256, H1=256, H2=128
#define PTS   65536        // B*N
#define K1C   384          // C1+C2
#define K2C   256          // H1
#define H1C   256
#define H2C   128

// ---------------- workspace layout (bytes) ----------------
#define OFF_SC  ((size_t)0)                         // s1[256] t1[256] s2[128] t2[128]
#define OFF_W   ((size_t)4096)                      // weights  PTS*3 f32
#define OFF_I   (OFF_W  + (size_t)PTS*3*4)          // indices  PTS*3 i32
#define OFF_XT  (OFF_I  + (size_t)PTS*3*4)          // X  bf16 [PTS][384]
#define OFF_Y1  (OFF_XT + (size_t)PTS*K1C*2)        // Y1 bf16 [PTS][256]
#define OFF_P1  (OFF_Y1 + (size_t)PTS*H1C*2)        // partials L1: 64*256*2 f32
#define OFF_P2  (OFF_P1 + (size_t)64*256*2*4)       // partials L2: 64*128*2 f32
#define OFF_WP1 (OFF_P2 + (size_t)64*128*2*4)       // W1 packed bf16 (256*384)
#define OFF_WP2 (OFF_WP1 + (size_t)H1C*K1C*2)       // W2 packed bf16 (128*256)
// Y2 f32 [PTS][128] aliases OFF_XT (X is dead after gemm1)

static __device__ __forceinline__ unsigned short f2bf(float f) {
  unsigned u = __builtin_bit_cast(unsigned, f);
  u += 0x7FFFu + ((u >> 16) & 1u);          // round-to-nearest-even
  return (unsigned short)(u >> 16);
}
static __device__ __forceinline__ float bf2f(unsigned short s) {
  return __builtin_bit_cast(float, (unsigned)s << 16);
}

// ---------------- 0) pack W (Cout x K, f32) into A-fragment bf16 layout ----------
// wp flat index = ((ot*(K/32) + kt)*32 + lane)*16 + j
// lane<16 holds row ot*16+lane,   K = kt*32 + {0..15}
// lane>=16 holds row ot*16+lane-16, K = kt*32 + {16..31}
__global__ __launch_bounds__(256) void packw_kernel(
    const float* __restrict__ W, unsigned short* __restrict__ wp, int K) {
  const int tid  = blockIdx.x * 256 + threadIdx.x;   // 0 .. Cout*K-1
  const int j    = tid & 15;
  const int lane = (tid >> 4) & 31;
  const int ktot = tid >> 9;
  const int KT   = K >> 5;
  const int kt   = ktot % KT;
  const int ot   = ktot / KT;
  const int h    = lane >> 4;
  const int row  = ot * 16 + (lane & 15);
  const int k    = kt * 32 + 16 * h + j;
  wp[tid] = f2bf(W[(size_t)row * K + k]);
}

// ---------------- 1) three_nn + inverse-distance weights ----------------
__global__ __launch_bounds__(256) void knn_kernel(
    const float* __restrict__ unknown, const float* __restrict__ known,
    float* __restrict__ wbuf, int* __restrict__ ibuf) {
  __shared__ float skx[4096], sky[4096], skz[4096];
  const int tid = threadIdx.x;
  const int gp  = blockIdx.x * 256 + tid;     // global point id (b*N+n)
  const int b   = gp >> 14;
  const float* kb = known + (size_t)b * 4096 * 3;
  for (int i = tid; i < 4096 * 3; i += 256) {
    float v = kb[i];
    int m = i / 3, k = i - m * 3;
    if (k == 0) skx[m] = v; else if (k == 1) sky[m] = v; else skz[m] = v;
  }
  __syncthreads();
  const float ux = unknown[gp*3+0], uy = unknown[gp*3+1], uz = unknown[gp*3+2];
  float d0 = 1e30f, d1 = 1e30f, d2 = 1e30f;
  int   i0 = 0, i1 = 0, i2 = 0;
  for (int m = 0; m < 4096; ++m) {
    float dx = ux - skx[m], dy = uy - sky[m], dz = uz - skz[m];
    float d = dx*dx + dy*dy + dz*dz;
    if (d < d2) {
      if (d < d1) {
        d2 = d1; i2 = i1;
        if (d < d0) { d1 = d0; i1 = i0; d0 = d; i0 = m; }
        else        { d1 = d;  i1 = m; }
      } else      { d2 = d;  i2 = m; }
    }
  }
  d0 = fmaxf(d0, 0.f); d1 = fmaxf(d1, 0.f); d2 = fmaxf(d2, 0.f);
  float r0 = 1.f/(d0+1e-8f), r1 = 1.f/(d1+1e-8f), r2 = 1.f/(d2+1e-8f);
  float rs = 1.f/(r0+r1+r2);
  wbuf[gp*3+0] = r0*rs; wbuf[gp*3+1] = r1*rs; wbuf[gp*3+2] = r2*rs;
  ibuf[gp*3+0] = i0;    ibuf[gp*3+1] = i1;    ibuf[gp*3+2] = i2;
}

// ---------------- 2) interpolate + concat -> X bf16 [p][384] ----------------
__global__ __launch_bounds__(128) void build_x_kernel(
    const float* __restrict__ kf, const float* __restrict__ uf,
    const float* __restrict__ wbuf, const int* __restrict__ ibuf,
    unsigned short* __restrict__ xt) {
  const int p = blockIdx.x;              // 0..65535
  const int t = threadIdx.x;             // 0..127
  const int b = p >> 14, n = p & 16383;
  const float w0 = wbuf[p*3+0], w1 = wbuf[p*3+1], w2 = wbuf[p*3+2];
  const int   i0 = ibuf[p*3+0], i1 = ibuf[p*3+1], i2 = ibuf[p*3+2];
  const float* kfb = kf + (size_t)b * 256 * 4096;
  unsigned short* xr = xt + (size_t)p * K1C;
  #pragma unroll
  for (int cc = 0; cc < 3; ++cc) {
    int c = t + cc * 128;
    float v;
    if (c < 256) {
      const float* row = kfb + (size_t)c * 4096;
      v = row[i0]*w0 + row[i1]*w1 + row[i2]*w2;
    } else {
      v = uf[((size_t)b * 128 + (c - 256)) * 16384 + n];
    }
    xr[c] = f2bf(v);
  }
}

// ---------------- 3) GEMM1: Y1 = W1(256x384) * X + b1, bf16 out ----------------
// wave computes 16 channels x 64 points (4 accumulators, 4x A reuse)
__global__ __launch_bounds__(256) void gemm1_kernel(
    const unsigned short* __restrict__ xt, const unsigned short* __restrict__ wp,
    const float* __restrict__ bias, unsigned short* __restrict__ y1) {
  const int lane = threadIdx.x & 31;
  const int wid  = blockIdx.x * 8 + (threadIdx.x >> 5);
  const int ot = wid & 15;               // 16 out-channel tiles
  const int pg = wid >> 4;               // 1024 groups of 64 points
  const int h  = lane >> 4, cl = lane & 15;
  v8f acc[4] = {{}, {}, {}, {}};
  const unsigned short* wl = wp + ((size_t)ot * (K1C/32)) * 512 + lane * 16;
  const unsigned short* xr = xt + ((size_t)(pg * 64 + cl)) * K1C + 16 * h;
  #pragma unroll
  for (int kt = 0; kt < K1C / 32; ++kt) {
    v16bf A = *reinterpret_cast<const v16bf*>(wl + kt * 512);
    #pragma unroll
    for (int q = 0; q < 4; ++q) {
      v16bf Bv = *reinterpret_cast<const v16bf*>(xr + (size_t)q * 16 * K1C + kt * 32);
      acc[q] = __builtin_amdgcn_wmma_f32_16x16x32_bf16(
          false, A, false, Bv, (short)0, acc[q], false, false);
    }
  }
  const int chBase = ot * 16 + 8 * h;
  float bv[8];
  #pragma unroll
  for (int i = 0; i < 8; ++i) bv[i] = bias[chBase + i];
  #pragma unroll
  for (int q = 0; q < 4; ++q) {
    const size_t p = (size_t)pg * 64 + q * 16 + cl;
    unsigned int pk[4];
    #pragma unroll
    for (int r = 0; r < 4; ++r) {
      float v0 = acc[q][2*r]   + bv[2*r];
      float v1 = acc[q][2*r+1] + bv[2*r+1];
      pk[r] = (unsigned)f2bf(v0) | ((unsigned)f2bf(v1) << 16);
    }
    *reinterpret_cast<uint4*>(y1 + p * H1C + chBase) = make_uint4(pk[0], pk[1], pk[2], pk[3]);
  }
}

// ---------------- 4) per-channel partial sums ----------------
__global__ __launch_bounds__(256) void stats_bf16_kernel(
    const unsigned short* __restrict__ y, float* __restrict__ psum, float* __restrict__ psq) {
  const int t = threadIdx.x, blk = blockIdx.x;       // 256 ch, 64 blocks x 1024 pts
  float s = 0.f, q = 0.f;
  const unsigned short* base = y + (size_t)blk * 1024 * 256 + t;
  for (int p = 0; p < 1024; ++p) { float v = bf2f(base[p * 256]); s += v; q += v * v; }
  psum[blk*256+t] = s; psq[blk*256+t] = q;
}
__global__ __launch_bounds__(128) void stats_f32_kernel(
    const float* __restrict__ y, float* __restrict__ psum, float* __restrict__ psq) {
  const int t = threadIdx.x, blk = blockIdx.x;       // 128 ch, 64 blocks x 1024 pts
  float s = 0.f, q = 0.f;
  const float* base = y + (size_t)blk * 1024 * 128 + t;
  for (int p = 0; p < 1024; ++p) { float v = base[p * 128]; s += v; q += v * v; }
  psum[blk*128+t] = s; psq[blk*128+t] = q;
}

// ---------------- 5) finalize BN: s = g*rsqrt(var+eps), t = be - mu*s ----------------
__global__ void finalize_kernel(const float* __restrict__ psum, const float* __restrict__ psq,
                                int nblk, int nch, const float* __restrict__ g,
                                const float* __restrict__ be,
                                float* __restrict__ sarr, float* __restrict__ tarr) {
  int c = blockIdx.x * blockDim.x + threadIdx.x;
  if (c >= nch) return;
  float s = 0.f, q = 0.f;
  for (int i = 0; i < nblk; ++i) { s += psum[i*nch+c]; q += psq[i*nch+c]; }
  const float inv = 1.f / (float)PTS;
  float mu  = s * inv;
  float var = q * inv - mu * mu;
  float rs  = rsqrtf(var + 1e-5f);
  float sc  = g[c] * rs;
  sarr[c] = sc; tarr[c] = be[c] - mu * sc;
}

// ---------------- 6) BN+ReLU in place on Y1 (bf16), feeds GEMM2 ----------------
__global__ __launch_bounds__(256) void bnrelu_kernel(
    unsigned short* __restrict__ y1, const float* __restrict__ s1, const float* __restrict__ t1) {
  size_t e = (size_t)blockIdx.x * 256 + threadIdx.x;   // uint4 index (8 bf16)
  uint4* ptr = reinterpret_cast<uint4*>(y1) + e;
  uint4 v = *ptr;
  const int c0 = (int)((e * 8) & 255);
  unsigned int w[4] = { v.x, v.y, v.z, v.w };
  #pragma unroll
  for (int q = 0; q < 4; ++q) {
    int c = c0 + 2 * q;
    float a = bf2f((unsigned short)(w[q] & 0xFFFFu));
    float b = bf2f((unsigned short)(w[q] >> 16));
    a = fmaxf(s1[c]   * a + t1[c],   0.f);
    b = fmaxf(s1[c+1] * b + t1[c+1], 0.f);
    w[q] = (unsigned)f2bf(a) | ((unsigned)f2bf(b) << 16);
  }
  *ptr = make_uint4(w[0], w[1], w[2], w[3]);
}

// ---------------- 7) GEMM2: Y2 = W2(128x256) * X2 + b2, f32 out ----------------
__global__ __launch_bounds__(256) void gemm2_kernel(
    const unsigned short* __restrict__ x2, const unsigned short* __restrict__ wp,
    const float* __restrict__ bias, float* __restrict__ y2) {
  const int lane = threadIdx.x & 31;
  const int wid  = blockIdx.x * 8 + (threadIdx.x >> 5);
  const int ot = wid & 7;                // 8 out-channel tiles
  const int pg = wid >> 3;               // 1024 groups of 64 points
  const int h  = lane >> 4, cl = lane & 15;
  v8f acc[4] = {{}, {}, {}, {}};
  const unsigned short* wl = wp + ((size_t)ot * (K2C/32)) * 512 + lane * 16;
  const unsigned short* xr = x2 + ((size_t)(pg * 64 + cl)) * K2C + 16 * h;
  #pragma unroll
  for (int kt = 0; kt < K2C / 32; ++kt) {
    v16bf A = *reinterpret_cast<const v16bf*>(wl + kt * 512);
    #pragma unroll
    for (int q = 0; q < 4; ++q) {
      v16bf Bv = *reinterpret_cast<const v16bf*>(xr + (size_t)q * 16 * K2C + kt * 32);
      acc[q] = __builtin_amdgcn_wmma_f32_16x16x32_bf16(
          false, A, false, Bv, (short)0, acc[q], false, false);
    }
  }
  const int chBase = ot * 16 + 8 * h;
  float bv[8];
  #pragma unroll
  for (int i = 0; i < 8; ++i) bv[i] = bias[chBase + i];
  #pragma unroll
  for (int q = 0; q < 4; ++q) {
    const size_t p = (size_t)pg * 64 + q * 16 + cl;
    float4 o0, o1;
    o0.x = acc[q][0] + bv[0]; o0.y = acc[q][1] + bv[1];
    o0.z = acc[q][2] + bv[2]; o0.w = acc[q][3] + bv[3];
    o1.x = acc[q][4] + bv[4]; o1.y = acc[q][5] + bv[5];
    o1.z = acc[q][6] + bv[6]; o1.w = acc[q][7] + bv[7];
    *reinterpret_cast<float4*>(y2 + p * H2C + chBase)     = o0;
    *reinterpret_cast<float4*>(y2 + p * H2C + chBase + 4) = o1;
  }
}

// ---------------- 8) final BN+ReLU + transpose to (B,128,N) ----------------
__global__ __launch_bounds__(256) void out_kernel(
    const float* __restrict__ y2, const float* __restrict__ s2,
    const float* __restrict__ t2, float* __restrict__ out) {
  __shared__ float tile[32][129];
  const int tid = threadIdx.x;
  const int p0  = blockIdx.x * 32;                 // 32 points per block
  const float* src = y2 + (size_t)p0 * H2C;
  for (int f = tid; f < 32 * 128; f += 256)
    tile[f >> 7][f & 127] = src[f];
  __syncthreads();
  const int w = tid >> 5, lane = tid & 31;
  const int b  = p0 >> 14;
  const int n0 = p0 & 16383;
  for (int c = w; c < 128; c += 8) {
    float v = tile[lane][c];
    v = fmaxf(s2[c] * v + t2[c], 0.f);
    out[((size_t)b * 128 + c) * 16384 + n0 + lane] = v;
  }
}

// ---------------- launch ----------------
extern "C" void kernel_launch(void* const* d_in, const int* in_sizes, int n_in,
                              void* d_out, int out_size, void* d_ws, size_t ws_size,
                              hipStream_t stream) {
  (void)in_sizes; (void)n_in; (void)out_size; (void)ws_size;
  const float* unknown = (const float*)d_in[0];
  const float* known   = (const float*)d_in[1];
  const float* uf      = (const float*)d_in[2];
  const float* kf      = (const float*)d_in[3];
  const float* W1      = (const float*)d_in[4];
  const float* b1      = (const float*)d_in[5];
  const float* g1      = (const float*)d_in[6];
  const float* be1     = (const float*)d_in[7];
  const float* W2      = (const float*)d_in[8];
  const float* b2      = (const float*)d_in[9];
  const float* g2      = (const float*)d_in[10];
  const float* be2     = (const float*)d_in[11];

  char* ws = (char*)d_ws;
  float* sc   = (float*)(ws + OFF_SC);
  float* s1   = sc;        float* t1 = sc + 256;
  float* s2   = sc + 512;  float* t2 = sc + 640;
  float* wbuf = (float*)(ws + OFF_W);
  int*   ibuf = (int*)  (ws + OFF_I);
  unsigned short* xt  = (unsigned short*)(ws + OFF_XT);
  unsigned short* y1  = (unsigned short*)(ws + OFF_Y1);
  unsigned short* wp1 = (unsigned short*)(ws + OFF_WP1);
  unsigned short* wp2 = (unsigned short*)(ws + OFF_WP2);
  float* y2  = (float*)(ws + OFF_XT);      // aliases X (dead after gemm1)
  float* p1s = (float*)(ws + OFF_P1); float* p1q = p1s + 64 * 256;
  float* p2s = (float*)(ws + OFF_P2); float* p2q = p2s + 64 * 128;
  float* out = (float*)d_out;

  packw_kernel     <<<(H1C * K1C) / 256, 256, 0, stream>>>(W1, wp1, K1C);
  packw_kernel     <<<(H2C * K2C) / 256, 256, 0, stream>>>(W2, wp2, K2C);
  knn_kernel       <<<PTS / 256, 256, 0, stream>>>(unknown, known, wbuf, ibuf);
  build_x_kernel   <<<PTS, 128, 0, stream>>>(kf, uf, wbuf, ibuf, xt);
  gemm1_kernel     <<<(16 * (PTS / 64)) / 8, 256, 0, stream>>>(xt, wp1, b1, y1);
  stats_bf16_kernel<<<64, 256, 0, stream>>>(y1, p1s, p1q);
  finalize_kernel  <<<1, 256, 0, stream>>>(p1s, p1q, 64, 256, g1, be1, s1, t1);
  bnrelu_kernel    <<<(PTS * H1C / 8) / 256, 256, 0, stream>>>(y1, s1, t1);
  gemm2_kernel     <<<(8 * (PTS / 64)) / 8, 256, 0, stream>>>(y1, wp2, b2, y2);
  stats_f32_kernel <<<64, 128, 0, stream>>>(y2, p2s, p2q);
  finalize_kernel  <<<1, 128, 0, stream>>>(p2s, p2q, 64, 128, g2, be2, s2, t2);
  out_kernel       <<<PTS / 32, 256, 0, stream>>>(y2, s2, t2, out);
}
